// SoftAgg_29678224016007
// MI455X (gfx1250) — matrix operand order repaired and checked by reading, not compiled, not run on hardware
//
#include <hip/hip_runtime.h>
#include <hip/hip_bf16.h>

// ---------------------------------------------------------------------------
// SoftAgg for MI455X (gfx1250): bf16 WMMA GEMMs with a 3-deep async-to-LDS
// pipeline + online segment softmax. Shape: B=4, N=16384, D=512, S=512.
// ---------------------------------------------------------------------------

#define USE_ASYNC_LDS_STAGE 1   // gfx1250 GLOBAL_LOAD_ASYNC_TO_LDS_B128 path

typedef __attribute__((ext_vector_type(16))) __bf16 v16bf;
typedef __attribute__((ext_vector_type(8)))  float  v8f;
typedef __attribute__((ext_vector_type(4)))  unsigned int v4u;

union Frag16 { v16bf bf; v4u q[2]; };

#define D_DIM 512
#define S_SEG 512
#define LDS_STRIDE 40   // 32 K-elems + 8 pad (ushorts) => 80B row stride

__device__ __forceinline__ unsigned short f32_to_bf16_rn(float f) {
    unsigned u = __float_as_uint(f);
    u += 0x7FFFu + ((u >> 16) & 1u);          // round-to-nearest-even
    return (unsigned short)(u >> 16);
}

#if USE_ASYNC_LDS_STAGE
// CDNA5 ISA: async load memory -> LDS, ASYNCcnt-tracked.
// VDST = LDS byte address (generic LDS pointer truncated to 32 bits per the
// ISA aperture rule), VADDR = 64-bit global address.
__device__ __forceinline__ void async_b128_to_lds(const unsigned short* gptr,
                                                  unsigned short* lptr) {
    unsigned lds_addr = (unsigned)(unsigned long long)(uintptr_t)lptr;
    unsigned long long gaddr = (unsigned long long)(uintptr_t)gptr;
    asm volatile("global_load_async_to_lds_b128 %0, %1, off"
                 :: "v"(lds_addr), "v"(gaddr) : "memory");
}
// Async loads retire in order (ISA 08 §4.1); each K-chunk is exactly 3 async
// ops per wave, so "<= 3" drains the oldest chunk while the newest stays in
// flight.
__device__ __forceinline__ void wait_asynccnt_3() {
    asm volatile("s_wait_asynccnt 3" ::: "memory");
}
__device__ __forceinline__ void wait_asynccnt_0() {
    asm volatile("s_wait_asynccnt 0" ::: "memory");
}
#endif

// ---------------------------------------------------------------------------
// K0: f32 -> bf16 conversion (vector-4)
// ---------------------------------------------------------------------------
__global__ __launch_bounds__(256) void cvt_f32_bf16_v4(
    const float4* __restrict__ src, ushort4* __restrict__ dst, long long n4) {
    long long i = (long long)blockIdx.x * blockDim.x + threadIdx.x;
    long long stride = (long long)gridDim.x * blockDim.x;
    for (; i < n4; i += stride) {
        float4 v = src[i];
        ushort4 o;
        o.x = f32_to_bf16_rn(v.x);
        o.y = f32_to_bf16_rn(v.y);
        o.z = f32_to_bf16_rn(v.z);
        o.w = f32_to_bf16_rn(v.w);
        dst[i] = o;
    }
}

// ---------------------------------------------------------------------------
// K1/K3: out[m, e] = sum_d A[m, d] * W[e, d] + bias[e]
// A: [M x 512] bf16 row-major, W: [512 x 512] bf16 row-major ([e][d]).
// Block: 256 threads = 8 waves; tile 64(M) x 128(E); wave tile 16 x 64.
// K-loop: 16 steps of K=32, 3-deep async LDS pipeline (chunks k+1 and k+2
// in flight while computing on chunk k).
// ---------------------------------------------------------------------------
__global__ __launch_bounds__(256) void gemm_bf16_wmma(
    const unsigned short* __restrict__ A,
    const unsigned short* __restrict__ W,
    const float* __restrict__ bias,
    float* __restrict__ out,
    int M) {
    __shared__ alignas(16) unsigned short sA[3][64  * LDS_STRIDE];  // 3 x 5KB
    __shared__ alignas(16) unsigned short sB[3][128 * LDS_STRIDE];  // 3 x 10KB

    const int t    = threadIdx.x;
    const int lane = t & 31;
    const int wid  = t >> 5;
    const int mw   = wid & 3;        // 4 waves along M
    const int nw   = wid >> 2;       // 2 waves along E
    const int r    = lane & 15;
    const int half = lane >> 4;

    const long long mrow = (long long)blockIdx.x * 64;
    const int ncol = blockIdx.y * 128;

    // Cooperative staging assignment (per K-chunk of 32):
    //   A chunk 64x32 : thread -> row t>>2, k-part (t&3)*8   (1 x b128)
    //   B chunk 128x32: thread -> col t>>1, k-part (t&1)*16  (2 x b128)
    const int a_row = t >> 2, a_kp = (t & 3) * 8;
    const int b_col = t >> 1, b_kp = (t & 1) * 16;
    const unsigned short* __restrict__ gA = A + (mrow + a_row) * (long long)D_DIM + a_kp;
    const unsigned short* __restrict__ gW = W + (long long)(ncol + b_col) * D_DIM + b_kp;
    const int aoff = a_row * LDS_STRIDE + a_kp;
    const int boff = b_col * LDS_STRIDE + b_kp;

    v8f acc[4];
#pragma unroll
    for (int j = 0; j < 4; ++j) acc[j] = (v8f)0.0f;

    // Fragment read bases in LDS (ushort units).
    const int afrag = (mw * 16 + r) * LDS_STRIDE + half * 8;      // + {0, 16}
    const int bfragbase = (nw * 64 + r) * LDS_STRIDE + half * 8;  // + j*16*LDS_STRIDE, {0, 16}

#if USE_ASYNC_LDS_STAGE
    // ---- prologue: chunks 0 and 1 in flight; drain chunk 0 ----
#pragma unroll
    for (int c = 0; c < 2; ++c) {
        async_b128_to_lds(gA + c * 32,     &sA[c][aoff]);
        async_b128_to_lds(gW + c * 32,     &sB[c][boff]);
        async_b128_to_lds(gW + c * 32 + 8, &sB[c][boff + 8]);
    }
    wait_asynccnt_3();
#else
    *(v4u*)(&sA[0][aoff])     = *(const v4u*)(gA);
    *(v4u*)(&sB[0][boff])     = *(const v4u*)(gW);
    *(v4u*)(&sB[0][boff + 8]) = *(const v4u*)(gW + 8);
#endif
    __syncthreads();

#pragma unroll
    for (int ki = 0; ki < 16; ++ki) {
        const int cur = ki % 3;

#if USE_ASYNC_LDS_STAGE
        // Keep two chunks in flight: issue chunk ki+2 into buffer (ki+2)%3.
        if (ki + 2 < 16) {
            const int pre = (ki + 2) % 3;
            const int kb2 = (ki + 2) * 32;
            async_b128_to_lds(gA + kb2,     &sA[pre][aoff]);
            async_b128_to_lds(gW + kb2,     &sB[pre][boff]);
            async_b128_to_lds(gW + kb2 + 8, &sB[pre][boff + 8]);
        }
#else
        v4u ra, rb0, rb1;
        if (ki < 15) {
            const int kb1 = (ki + 1) * 32;
            ra  = *(const v4u*)(gA + kb1);
            rb0 = *(const v4u*)(gW + kb1);
            rb1 = *(const v4u*)(gW + kb1 + 8);
        }
#endif

        // Fragments from LDS (ds_load_b128), then 4 WMMAs.
        Frag16 a;
        {
            const unsigned short* pa = &sA[cur][afrag];
            a.q[0] = *(const v4u*)(pa);
            a.q[1] = *(const v4u*)(pa + 16);
        }
#pragma unroll
        for (int j = 0; j < 4; ++j) {
            Frag16 b;
            const unsigned short* pb = &sB[cur][bfragbase + j * 16 * LDS_STRIDE];
            b.q[0] = *(const v4u*)(pb);
            b.q[1] = *(const v4u*)(pb + 16);
            acc[j] = __builtin_amdgcn_wmma_f32_16x16x32_bf16(
                false, a.bf, false, b.bf, (short)0, acc[j], false, false);
        }

        if (ki < 15) {
#if USE_ASYNC_LDS_STAGE
            // Drain chunk ki+1 (leave ki+2 in flight when it exists).
            if (ki < 14) wait_asynccnt_3(); else wait_asynccnt_0();
#else
            const int nxt = (ki + 1) % 3;
            *(v4u*)(&sA[nxt][aoff])     = ra;
            *(v4u*)(&sB[nxt][boff])     = rb0;
            *(v4u*)(&sB[nxt][boff + 8]) = rb1;
#endif
            __syncthreads();
        }
    }

    // Epilogue: C layout (32-bit 16x16): lane -> col N = lane&15;
    // VGPR v -> row M = v + (lane>=16 ? 8 : 0).
    const long long rbase = mrow + mw * 16 + half * 8;
#pragma unroll
    for (int j = 0; j < 4; ++j) {
        const int c = ncol + nw * 64 + j * 16 + r;
        const float bv = bias[c];
#pragma unroll
        for (int v = 0; v < 8; ++v) {
            out[(rbase + v) * (long long)D_DIM + c] = acc[j][v] + bv;
        }
    }
    (void)M;
}

// ---------------------------------------------------------------------------
// K2: per-(b, s) online segment softmax over gx + weighted sum of fx -> y(bf16)
// ---------------------------------------------------------------------------
__device__ __forceinline__ int lower_bound_i32(const int* __restrict__ a, int n, int v) {
    int lo = 0, hi = n;
    while (lo < hi) {
        int mid = (lo + hi) >> 1;
        if (a[mid] < v) lo = mid + 1; else hi = mid;
    }
    return lo;
}

__global__ __launch_bounds__(256) void seg_softmax_agg(
    const float* __restrict__ fx, const float* __restrict__ gx,
    const int* __restrict__ jx, unsigned short* __restrict__ ybf, int N) {
    const int s = blockIdx.x;
    const int b = blockIdx.y;
    const int n0 = lower_bound_i32(jx, N, s);
    const int n1 = lower_bound_i32(jx, N, s + 1);

    const int d = threadIdx.x;           // columns d and d+256
    float m0 = -3.4e38f, den0 = 0.f, num0 = 0.f;
    float m1 = -3.4e38f, den1 = 0.f, num1 = 0.f;

    const float* __restrict__ g = gx + ((long long)b * N + n0) * D_DIM;
    const float* __restrict__ f = fx + ((long long)b * N + n0) * D_DIM;
    for (int n = n0; n < n1; ++n, g += D_DIM, f += D_DIM) {
        float g0 = g[d], g1 = g[d + 256];
        float f0 = f[d], f1 = f[d + 256];
        float nm0 = fmaxf(m0, g0), nm1 = fmaxf(m1, g1);
        float sc0 = __expf(m0 - nm0), sc1 = __expf(m1 - nm1);
        float e0  = __expf(g0 - nm0), e1  = __expf(g1 - nm1);
        den0 = den0 * sc0 + e0;       den1 = den1 * sc1 + e1;
        num0 = num0 * sc0 + f0 * e0;  num1 = num1 * sc1 + f1 * e1;
        m0 = nm0;                     m1 = nm1;
    }
    float y0 = (n1 > n0) ? (num0 / den0) : 0.f;
    float y1 = (n1 > n0) ? (num1 / den1) : 0.f;
    unsigned short* __restrict__ yo = ybf + ((long long)b * S_SEG + s) * D_DIM;
    yo[d]       = f32_to_bf16_rn(y0);
    yo[d + 256] = f32_to_bf16_rn(y1);
}

// ---------------------------------------------------------------------------
// K4: out[b, n, :] = hy[b, jx[n], :]   (float4-coalesced, hy lives in L2)
// ---------------------------------------------------------------------------
__global__ __launch_bounds__(128) void gather_rows(
    const float* __restrict__ hy, const int* __restrict__ jx,
    float* __restrict__ out, int N) {
    const int n = blockIdx.x;
    const int b = blockIdx.y;
    const int s = jx[n];
    const float4* __restrict__ src = (const float4*)(hy + ((long long)b * S_SEG + s) * D_DIM);
    float4* __restrict__ dst = (float4*)(out + ((long long)b * N + n) * D_DIM);
    dst[threadIdx.x] = src[threadIdx.x];   // 128 threads * 16B = 2KB row
}

// ---------------------------------------------------------------------------
// Host-side orchestration
// ---------------------------------------------------------------------------
extern "C" void kernel_launch(void* const* d_in, const int* in_sizes, int n_in,
                              void* d_out, int out_size, void* d_ws, size_t ws_size,
                              hipStream_t stream) {
    const float* x   = (const float*)d_in[0];
    const int*   jx  = (const int*)  d_in[1];
    // d_in[2] = num_segments scalar (S=512, fixed by the reference shape)
    const float* Wf  = (const float*)d_in[3];
    const float* bf_ = (const float*)d_in[4];
    const float* Wg  = (const float*)d_in[5];
    const float* bg  = (const float*)d_in[6];
    const float* Wh  = (const float*)d_in[7];
    const float* bh  = (const float*)d_in[8];

    const int N = in_sizes[1];                                   // 16384
    const int D = in_sizes[4];                                   // 512
    const long long total = (long long)in_sizes[0];
    const int B = (int)(total / ((long long)N * D));             // 4
    const long long MR = (long long)B * N;                       // 65536
    const int S = S_SEG;                                         // 512
    const int MS = B * S;                                        // 2048

    size_t off = 0;
    auto alloc = [&](size_t bytes) -> void* {
        off = (off + 255) & ~(size_t)255;
        void* p = (char*)d_ws + off;
        off += bytes;
        return p;
    };
    unsigned short* xbf = (unsigned short*)alloc((size_t)MR * D * 2);
    unsigned short* wfb = (unsigned short*)alloc((size_t)D * D * 2);
    unsigned short* wgb = (unsigned short*)alloc((size_t)D * D * 2);
    unsigned short* whb = (unsigned short*)alloc((size_t)D * D * 2);
    float* fx = (float*)alloc((size_t)MR * D * 4);
    float* gx = (float*)alloc((size_t)MR * D * 4);
    unsigned short* ybf = (unsigned short*)alloc((size_t)MS * D * 2);
    float* hy = (float*)alloc((size_t)MS * D * 4);
    (void)ws_size; (void)n_in; (void)out_size;

    // K0: conversions to bf16
    {
        long long n4x = (MR * (long long)D) / 4;
        cvt_f32_bf16_v4<<<8192, 256, 0, stream>>>((const float4*)x, (ushort4*)xbf, n4x);
        long long n4w = ((long long)D * D) / 4;
        cvt_f32_bf16_v4<<<256, 256, 0, stream>>>((const float4*)Wf, (ushort4*)wfb, n4w);
        cvt_f32_bf16_v4<<<256, 256, 0, stream>>>((const float4*)Wg, (ushort4*)wgb, n4w);
        cvt_f32_bf16_v4<<<256, 256, 0, stream>>>((const float4*)Wh, (ushort4*)whb, n4w);
    }

    // K1: fx = x*Wf^T + bf ; gx = x*Wg^T + bg
    {
        dim3 grid((unsigned)(MR / 64), (unsigned)(D / 128));
        gemm_bf16_wmma<<<grid, 256, 0, stream>>>(xbf, wfb, bf_, fx, (int)MR);
        gemm_bf16_wmma<<<grid, 256, 0, stream>>>(xbf, wgb, bg, gx, (int)MR);
    }

    // K2: segment softmax + weighted segment sum -> y (bf16)
    {
        dim3 grid((unsigned)S, (unsigned)B);
        seg_softmax_agg<<<grid, 256, 0, stream>>>(fx, gx, jx, ybf, N);
    }

    // K3: hy = y*Wh^T + bh
    {
        dim3 grid((unsigned)(MS / 64), (unsigned)(D / 128));
        gemm_bf16_wmma<<<grid, 256, 0, stream>>>(ybf, whb, bh, hy, MS);
    }

    // K4: out = hy[:, jx]
    {
        dim3 grid((unsigned)N, (unsigned)B);
        gather_rows<<<grid, 128, 0, stream>>>(hy, jx, (float*)d_out, N);
    }
}
